// CPCLayer_69355131896482
// MI455X (gfx1250) — matrix operand order repaired and checked by reading, not compiled, not run on hardware
//
#include <hip/hip_runtime.h>
#include <hip/hip_bf16.h>

typedef __attribute__((ext_vector_type(16))) _Float16 v16h;
typedef __attribute__((ext_vector_type(8)))  _Float16 v8h;
typedef __attribute__((ext_vector_type(4)))  _Float16 v4h;
typedef __attribute__((ext_vector_type(8)))  float    v8f;

union H16 { v16h v; v8h h[2]; };

// Problem constants
#define BB 1024      // batch
#define TT 12        // timesteps
#define DD 512       // feature dim
#define RS (TT*DD)   // row stride in elements (6144)

// Tiling: WG = 512 rows (4 i-blocks of 128) x 128 cols; 4 waves of 64x64
#define BM 128       // rows per i-block
#define BN 128       // columns per workgroup
#define NIB 4        // i-blocks per WG (512 rows)
#define LDA 520      // padded LDS row stride (halves); 1040B rows: 16B aligned, bank-skewed
#define LDP 520

// LDS layout (bytes)
#define OFF_A   0
#define OFF_P   133120           // 128*520*2
#define OFF_WMX 266240           // + 128*520*2
#define OFF_WSM 267264           // + 2*128*4
#define OFF_M   268288           // + 2*128*4
#define OFF_S   268800           // + 128*4
#define OFF_DG  269312           // + 128*4
#define SMEM_SZ 269824           // + 128*4

// Workspace layout (floats)
#define WS_M 0                   // [12][2][1024] per-half column max
#define WS_S 24576               // [12][2][1024] per-half column sumexp
#define WS_D 49152               // [12][1024]    diagonal values

__global__ __launch_bounds__(128, 1)
void cpc_fused_gemm_softmax(const float* __restrict__ P,   // predictions  [B,T,D] -> columns j
                            const float* __restrict__ X,   // x_future_enc [B,T,D] -> rows i
                            float* __restrict__ wsM,
                            float* __restrict__ wsS,
                            float* __restrict__ wsD) {
    extern __shared__ char smem[];
    _Float16* sA    = (_Float16*)(smem + OFF_A);
    _Float16* sP    = (_Float16*)(smem + OFF_P);
    float* sWaveMax = (float*)(smem + OFF_WMX);   // [2 rowgroups][128]
    float* sWaveSum = (float*)(smem + OFF_WSM);   // [2 rowgroups][128]
    float* sM       = (float*)(smem + OFF_M);     // running col max [128]
    float* sS       = (float*)(smem + OFF_S);     // running col sumexp [128]
    float* sDiag    = (float*)(smem + OFF_DG);    // diag value per col [128]

    const int tid  = threadIdx.x;        // 0..127 (4 waves)
    const int w    = tid >> 5;
    const int lane = tid & 31;
    const int half = lane >> 4;          // 0: lanes 0-15, 1: lanes 16-31
    const int m    = lane & 15;
    const int rw   = w >> 1;             // wave row group (0/1): rows rw*64..+64
    const int cw   = w & 1;              // wave col group (0/1): cols cw*64..+64

    const int jt    = blockIdx.x;        // 8 column tiles
    const int t     = blockIdx.y;        // 12 timesteps
    const int ih    = blockIdx.z;        // row half (0: rows 0-511, 1: rows 512-1023)
    const int jbase = jt * BN;

    const float* gP  = P + (size_t)jbase * RS + (size_t)t * DD;
    const float* gXt = X + (size_t)t * DD + (size_t)(ih * 512) * RS;

    // ---- stage P stripe (128 x 512) as f16 into LDS, once ----
    #pragma unroll 4
    for (int idx = tid; idx < BN * (DD / 4); idx += 128) {
        int r  = idx >> 7;               // row 0..127
        int c4 = idx & 127;              // float4 column
        float4 f = *(const float4*)(gP + (size_t)r * RS + c4 * 4);
        v4h hv = { (_Float16)f.x, (_Float16)f.y, (_Float16)f.z, (_Float16)f.w };
        *(v4h*)(sP + r * LDP + c4 * 4) = hv;
    }
    if (tid < BN) {
        sM[tid] = -3.0e38f;
        sS[tid] = 0.0f;
        sDiag[tid] = 0.0f;
    }

    // the diagonal rows for columns [jbase, jbase+128) form global i-block jt
    const bool diagHalf = (ih == (jt >> 2));
    const int  diagIb   = jt & 3;

    for (int ib = 0; ib < NIB; ++ib) {
        __syncthreads();                 // prev-iter LDS consumers done; sA free

        // ---- stage A block (128 x 512) as f16 into LDS ----
        const float* gA = gXt + (size_t)(ib * BM) * RS;
        #pragma unroll 4
        for (int idx = tid; idx < BM * (DD / 4); idx += 128) {
            int r  = idx >> 7;
            int c4 = idx & 127;
            float4 f = *(const float4*)(gA + (size_t)r * RS + c4 * 4);
            v4h hv = { (_Float16)f.x, (_Float16)f.y, (_Float16)f.z, (_Float16)f.w };
            *(v4h*)(sA + r * LDA + c4 * 4) = hv;
        }
        __syncthreads();

        // ---- GEMM: wave owns rows [rw*64, rw*64+64) x cols [cw*64, cw*64+64) ----
        v8f acc[4][4] = {};
        #pragma unroll 2
        for (int kb = 0; kb < DD / 32; ++kb) {
            const int k0 = kb * 32;
            H16 a[4], b[4];
            #pragma unroll
            for (int rt = 0; rt < 4; ++rt) {
                const _Float16* pa = sA + (rw * 64 + rt * 16 + m) * LDA + k0 + half * 8;
                a[rt].h[0] = *(const v8h*)pa;          // K 0-7   (lanes 0-15) / K 8-15  (lanes 16-31)
                a[rt].h[1] = *(const v8h*)(pa + 16);   // K 16-23 (lanes 0-15) / K 24-31 (lanes 16-31)
            }
            #pragma unroll
            for (int ct = 0; ct < 4; ++ct) {
                const _Float16* pb = sP + (cw * 64 + ct * 16 + m) * LDP + k0 + half * 8;
                b[ct].h[0] = *(const v8h*)pb;          // K 0-7   / K 8-15
                b[ct].h[1] = *(const v8h*)(pb + 16);   // K 16-23 / K 24-31
            }
            #pragma unroll
            for (int rt = 0; rt < 4; ++rt)
                #pragma unroll
                for (int ct = 0; ct < 4; ++ct)
                    acc[rt][ct] = __builtin_amdgcn_wmma_f32_16x16x32_f16(
                        false, a[rt].v, false, b[ct].v,
                        (short)0, acc[rt][ct], false, false);
        }

        // ---- pass (a): per-column max over this wave's 64 rows ----
        #pragma unroll
        for (int ct = 0; ct < 4; ++ct) {
            float v = -3.0e38f;
            #pragma unroll
            for (int rt = 0; rt < 4; ++rt)
                #pragma unroll
                for (int r = 0; r < 8; ++r)
                    v = fmaxf(v, acc[rt][ct][r]);
            v = fmaxf(v, __shfl_xor(v, 16, 32));
            if (lane < 16) sWaveMax[rw * 128 + cw * 64 + ct * 16 + m] = v;
        }
        // diagonal capture (only the i-block whose rows overlap our columns)
        if (diagHalf && ib == diagIb) {
            #pragma unroll
            for (int rt = 0; rt < 4; ++rt)
                #pragma unroll
                for (int ct = 0; ct < 4; ++ct) {
                    const int lcol  = cw * 64 + ct * 16 + m;     // local col == local row of diag
                    const int rbase = rw * 64 + rt * 16 + half * 8;
                    #pragma unroll
                    for (int r = 0; r < 8; ++r)
                        if (rbase + r == lcol) sDiag[lcol] = acc[rt][ct][r];
                }
        }
        __syncthreads();

        // ---- pass (b): sum of exp(val - blockmax) over this i-block ----
        #pragma unroll
        for (int ct = 0; ct < 4; ++ct) {
            const int c = cw * 64 + ct * 16 + m;
            float bm = fmaxf(sWaveMax[c], sWaveMax[128 + c]);
            float es = 0.0f;
            #pragma unroll
            for (int rt = 0; rt < 4; ++rt)
                #pragma unroll
                for (int r = 0; r < 8; ++r)
                    es += __expf(acc[rt][ct][r] - bm);
            es += __shfl_xor(es, 16, 32);
            if (lane < 16) sWaveSum[rw * 128 + c] = es;
        }
        __syncthreads();

        // ---- merge block stats into running (m, s) — wave 0 only ----
        if (w == 0) {
            #pragma unroll
            for (int c = lane; c < BN; c += 32) {
                float bm = fmaxf(sWaveMax[c], sWaveMax[128 + c]);
                float bs = sWaveSum[c] + sWaveSum[128 + c];
                float mr = sM[c], sr = sS[c];
                if (bm > mr) { sr = sr * __expf(mr - bm) + bs; mr = bm; }
                else         { sr += bs * __expf(bm - mr); }
                sM[c] = mr; sS[c] = sr;
            }
        }
    }
    __syncthreads();

    // ---- epilogue: write per-half column stats (+ diag) to workspace ----
    if (w == 0) {
        #pragma unroll
        for (int c = lane; c < BN; c += 32) {
            const int jc = jbase + c;
            wsM[(t * 2 + ih) * BB + jc] = sM[c];
            wsS[(t * 2 + ih) * BB + jc] = sS[c];
            if (diagHalf) wsD[t * BB + jc] = sDiag[c];
        }
    }
}

__global__ void cpc_finalize(const float* __restrict__ wsM,
                             const float* __restrict__ wsS,
                             const float* __restrict__ wsD,
                             float* __restrict__ out) {
    __shared__ float sl[256];
    __shared__ float sc[256];
    const int tid = threadIdx.x;
    float lsum = 0.0f, csum = 0.0f;
    for (int idx = tid; idx < TT * BB; idx += 256) {
        const int t = idx >> 10;
        const int j = idx & (BB - 1);
        float m0 = wsM[(t * 2 + 0) * BB + j];
        float m1 = wsM[(t * 2 + 1) * BB + j];
        float s0 = wsS[(t * 2 + 0) * BB + j];
        float s1 = wsS[(t * 2 + 1) * BB + j];
        float mm = fmaxf(m0, m1);
        float ss = s0 * __expf(m0 - mm) + s1 * __expf(m1 - mm);
        float lse = mm + __logf(ss);
        float dg = wsD[t * BB + j];
        lsum += dg - lse;
        csum += (dg >= mm) ? 1.0f : 0.0f;
    }
    sl[tid] = lsum; sc[tid] = csum;
    __syncthreads();
    #pragma unroll
    for (int s = 128; s > 0; s >>= 1) {
        if (tid < s) { sl[tid] += sl[tid + s]; sc[tid] += sc[tid + s]; }
        __syncthreads();
    }
    if (tid == 0) {
        const float inv = 1.0f / (float)(TT * BB);
        out[0] = -sl[0] * inv;   // -loss
        out[1] = sc[0] * inv;    // accuracy
    }
}

extern "C" void kernel_launch(void* const* d_in, const int* in_sizes, int n_in,
                              void* d_out, int out_size, void* d_ws, size_t ws_size,
                              hipStream_t stream) {
    const float* P = (const float*)d_in[0];   // predictions   -> GEMM columns
    const float* X = (const float*)d_in[1];   // x_future_enc  -> GEMM rows
    float* wsbase = (float*)d_ws;
    float* wsM = wsbase + WS_M;
    float* wsS = wsbase + WS_S;
    float* wsD = wsbase + WS_D;
    float* out = (float*)d_out;

    dim3 grid(8, 12, 2);
    cpc_fused_gemm_softmax<<<grid, 128, SMEM_SZ, stream>>>(P, X, wsM, wsS, wsD);
    cpc_finalize<<<1, 256, 0, stream>>>(wsM, wsS, wsD, out);
}
